// SpeakerAttention_50818053046799
// MI455X (gfx1250) — matrix operand order repaired and checked by reading, not compiled, not run on hardware
//
#include <hip/hip_runtime.h>
#include <hip/hip_bf16.h>

typedef _Float16 h16;
typedef __attribute__((ext_vector_type(16))) _Float16 v16h;
typedef __attribute__((ext_vector_type(8)))  _Float16 v8h;
typedef __attribute__((ext_vector_type(8)))  float    v8f;

// Problem dims (fixed by reference setup_inputs)
constexpr int Bn = 16;    // batch
constexpr int Cc = 512;   // channels
constexpr int Tt = 1024;  // time (Tc == Ts)

// ---------------------------------------------------------------------------
// WMMA fragment helpers (layouts per CDNA5 ISA §7.12.2, wave32)
// ---------------------------------------------------------------------------

// A fragment: 16x32 f16, source row-major (M x K), ld in halfs.
__device__ __forceinline__ v16h load_a16x32(const h16* tile, int ld) {
    const int lane = threadIdx.x & 31;
    const int m  = lane & 15;
    const int kh = (lane >> 4) << 3;            // 0 or 8
    const h16* r = tile + m * ld + kh;
    v8h lo = *reinterpret_cast<const v8h*>(r);        // K = kh .. kh+7
    v8h hi = *reinterpret_cast<const v8h*>(r + 16);   // K = kh+16 .. kh+23
    v16h out;
#pragma unroll
    for (int i = 0; i < 8; ++i) { out[i] = lo[i]; out[i + 8] = hi[i]; }
    return out;
}

// B fragment: 32x16 f16. Source is B^T stored row-major: (N rows) x (K).
__device__ __forceinline__ v16h load_b32x16(const h16* tileT, int ld) {
    const int lane = threadIdx.x & 31;
    const int n  = lane & 15;
    const int kh = (lane >> 4) << 3;            // 0 or 8
    const h16* r = tileT + n * ld + kh;
    v8h lo = *reinterpret_cast<const v8h*>(r);
    v8h hi = *reinterpret_cast<const v8h*>(r + 16);
    v16h out;
#pragma unroll
    for (int i = 0; i < 8; ++i) { out[i] = lo[i]; out[i + 8] = hi[i]; }
    return out;
}

__device__ __forceinline__ v8f wmma32(v16h a, v16h b, v8f c) {
    return __builtin_amdgcn_wmma_f32_16x16x32_f16(false, a, false, b, (short)0, c,
                                                  false, false);
}

// ---------------------------------------------------------------------------
// CDNA5 async global->LDS copy (ASYNCcnt-tracked), CDNA5 ISA
// GV mode: 64-bit global address in VGPR pair, LDS byte offset in VDST VGPR.
// ---------------------------------------------------------------------------
__device__ __forceinline__ void async_g2l_b128(unsigned int lds_byte_off, const void* g) {
    asm volatile("global_load_async_to_lds_b128 %0, %1, off"
                 :: "v"(lds_byte_off), "v"((unsigned long long)(uintptr_t)g)
                 : "memory");
}
__device__ __forceinline__ void wait_async0() {
    asm volatile("s_wait_asynccnt 0x0" ::: "memory");
}

// ---------------------------------------------------------------------------
// Kernel 1: fp32 -> f16 weight conversion (4 x 512x512)
// ---------------------------------------------------------------------------
__global__ void cvt_weights_kernel(const float* __restrict__ Wf, const float* __restrict__ Wg,
                                   const float* __restrict__ Wh, const float* __restrict__ Wo,
                                   h16* __restrict__ WfH, h16* __restrict__ WgH,
                                   h16* __restrict__ WhH, h16* __restrict__ WoH) {
    const int i = blockIdx.x * blockDim.x + threadIdx.x;
    const float* src; h16* dst;
    switch (blockIdx.y) {
        case 0:  src = Wf; dst = WfH; break;
        case 1:  src = Wg; dst = WgH; break;
        case 2:  src = Wh; dst = WhH; break;
        default: src = Wo; dst = WoH; break;
    }
    dst[i] = (h16)src[i];
}

// ---------------------------------------------------------------------------
// Kernel 2: mean-variance norm over time (unbiased var + eps), cast to f16,
// write TRANSPOSED (B,T,C).  blockIdx.y: 0 = content, 1 = style (+ raw copy)
// ---------------------------------------------------------------------------
__global__ void mvn_cast_kernel(const float* __restrict__ content,
                                const float* __restrict__ style,
                                h16* __restrict__ XcT, h16* __restrict__ XsT,
                                h16* __restrict__ XsR) {
    const int row   = blockIdx.x;          // b*C + c
    const int which = blockIdx.y;
    const int b = row / Cc;
    const int c = row % Cc;
    const float* src = (which == 0 ? content : style) + (size_t)row * Tt;

    __shared__ float s1[256];
    __shared__ float s2[256];
    const int tid = threadIdx.x;

    float x[4], sum = 0.f, sq = 0.f;
#pragma unroll
    for (int i = 0; i < 4; ++i) {
        x[i] = src[tid + i * 256];
        sum += x[i];
        sq  += x[i] * x[i];
    }
    s1[tid] = sum; s2[tid] = sq;
    __syncthreads();
    for (int s = 128; s > 0; s >>= 1) {
        if (tid < s) { s1[tid] += s1[tid + s]; s2[tid] += s2[tid + s]; }
        __syncthreads();
    }
    const float m    = s1[0] * (1.0f / Tt);
    const float var  = (s2[0] - (float)Tt * m * m) * (1.0f / (Tt - 1));
    const float rinv = rsqrtf(var + 1e-5f);

    h16* dstN = (which == 0 ? XcT : XsT) + (size_t)b * Tt * Cc + c;
#pragma unroll
    for (int i = 0; i < 4; ++i) {
        const int t = tid + i * 256;
        dstN[(size_t)t * Cc] = (h16)((x[i] - m) * rinv);
        if (which == 1)
            XsR[(size_t)b * Tt * Cc + (size_t)t * Cc + c] = (h16)x[i];
    }
}

// ---------------------------------------------------------------------------
// Kernel 3/5: conv1x1 as WMMA GEMM. Y = W(f16) @ X + bias, per batch.
//   A = W (Cout x Cin) row-major;  B^T = XT (B, T, C) row-major.
// Wave tile: 32 (cout) x 64 (t)  -> 8 WMMAs per 12 B128 loads per K-step.
// MODE 0: f16 Y (B,C,T)  | MODE 1: f16 Y^T (B,T,C) | MODE 2: f32 Y (B,C,T)
// ---------------------------------------------------------------------------
template <int MODE>
__global__ void conv1x1_gemm_kernel(const h16* __restrict__ W, const float* __restrict__ bias,
                                    const h16* __restrict__ XT,
                                    h16* __restrict__ Yh, float* __restrict__ Yf) {
    const int wave = threadIdx.x >> 5;
    const int tile = blockIdx.x * 8 + wave;
    constexpr int nTiles = Tt / 64;                   // 16
    constexpr int tilesPerBatch = (Cc / 32) * nTiles; // 256
    const int b  = tile / tilesPerBatch;
    const int r  = tile % tilesPerBatch;
    const int m0 = (r / nTiles) * 32;   // cout tile base
    const int n0 = (r % nTiles) * 64;   // t tile base

    const h16* A0 = W + (size_t)m0 * Cc;
    const h16* A1 = W + (size_t)(m0 + 16) * Cc;
    const h16* Bb = XT + (size_t)b * Tt * Cc + (size_t)n0 * Cc;

    v8f zero = {};
    v8f acc[2][4];
#pragma unroll
    for (int mi = 0; mi < 2; ++mi)
#pragma unroll
        for (int nt = 0; nt < 4; ++nt) acc[mi][nt] = zero;

#pragma unroll 2
    for (int kk = 0; kk < Cc; kk += 32) {
        v16h a0 = load_a16x32(A0 + kk, Cc);
        v16h a1 = load_a16x32(A1 + kk, Cc);
#pragma unroll
        for (int nt = 0; nt < 4; ++nt) {
            v16h bb = load_b32x16(Bb + (size_t)(nt * 16) * Cc + kk, Cc);
            acc[0][nt] = wmma32(a0, bb, acc[0][nt]);
            acc[1][nt] = wmma32(a1, bb, acc[1][nt]);
        }
    }

    const int lane = threadIdx.x & 31;
    const int n  = lane & 15;
    const int mb = (lane >> 4) * 8;
#pragma unroll
    for (int mi = 0; mi < 2; ++mi)
#pragma unroll
        for (int nt = 0; nt < 4; ++nt)
#pragma unroll
            for (int i = 0; i < 8; ++i) {
                const int m  = m0 + mi * 16 + mb + i;
                const int nn = n0 + nt * 16 + n;
                const float v = acc[mi][nt][i] + bias[m];
                if (MODE == 0)
                    Yh[((size_t)b * Cc + m) * Tt + nn] = (h16)v;
                else if (MODE == 1)
                    Yh[((size_t)b * Tt + nn) * Cc + m] = (h16)v;
                else
                    Yf[((size_t)b * Cc + m) * Tt + nn] = v;
            }
}

// ---------------------------------------------------------------------------
// Kernel 4: fused attention. One block = (b, 16-query tile), 8 waves.
//  Phase 0: async-stage the 16x512 Fq query tile into LDS
//           (global_load_async_to_lds_b128 + s_wait_asynccnt).
//  Phase 1: S(16x1024) = Aq(LDS) @ Gk (8 waves x 128 keys, WMMA) -> LDS f16.
//  Phase 2: row softmax in LDS (v_exp_f32).
//  Phase 3: O^T(16x512) = P(LDS) @ Hv^T (8 waves x 64 channels, WMMA).
// ---------------------------------------------------------------------------
__global__ void attn_fused_kernel(const h16* __restrict__ FqT,  // (B,Tc,C)
                                  const h16* __restrict__ GkT,  // (B,Ts,C)
                                  const h16* __restrict__ Hv,   // (B,C,Ts)
                                  h16* __restrict__ OT) {       // (B,Tc,C)
    constexpr int LDP = Tt + 8;                    // prob row stride (halfs)
    constexpr int LDA = Cc + 8;                    // query row stride (halfs)
    __shared__ __align__(16) h16 P[16 * LDP];      // 33 KB logits/probs
    __shared__ __align__(16) h16 Aq[16 * LDA];     // 16.6 KB staged Fq tile
    __shared__ float red[16][17];
    __shared__ float rowstat[16];

    const int b    = blockIdx.x / (Tt / 16);
    const int tq0  = (blockIdx.x % (Tt / 16)) * 16;
    const int wave = threadIdx.x >> 5;
    const int lane = threadIdx.x & 31;

    const h16* Asrc = FqT + ((size_t)b * Tt + tq0) * Cc;  // 16 query rows x C
    const h16* Gb   = GkT + (size_t)b * Tt * Cc;          // (s, c) rows
    v8f zero = {};

    // ---- Phase 0: async-stage Fq tile into LDS (1024 x b128 chunks) ----
    {
        const unsigned int aq_base = (unsigned int)(uintptr_t)&Aq[0];
#pragma unroll
        for (int i = 0; i < 4; ++i) {
            const int idx  = threadIdx.x + i * 256;   // 0..1023
            const int rrow = idx >> 6;                // /64 chunks per row
            const int c8   = (idx & 63) << 3;         // half offset in row
            async_g2l_b128(aq_base + (unsigned int)(rrow * LDA + c8) * 2,
                           Asrc + (size_t)rrow * Cc + c8);
        }
        wait_async0();
    }
    __syncthreads();

    // ---- Phase 1: logits (A from LDS, shared by all 8 waves) ----
    {
        v8f acc[8];
#pragma unroll
        for (int i = 0; i < 8; ++i) acc[i] = zero;
        const int s0 = wave * 128;
        for (int kk = 0; kk < Cc; kk += 32) {
            v16h a = load_a16x32(&Aq[0] + kk, LDA);   // ds_load_b128
#pragma unroll
            for (int nt = 0; nt < 8; ++nt) {
                v16h bb = load_b32x16(Gb + (size_t)(s0 + nt * 16) * Cc + kk, Cc);
                acc[nt] = wmma32(a, bb, acc[nt]);
            }
        }
        const int n  = lane & 15;
        const int mb = (lane >> 4) * 8;
#pragma unroll
        for (int nt = 0; nt < 8; ++nt)
#pragma unroll
            for (int i = 0; i < 8; ++i)
                P[(mb + i) * LDP + s0 + nt * 16 + n] = (h16)acc[nt][i];
    }
    __syncthreads();

    // ---- Phase 2: softmax over 1024 keys per row (16 threads/row) ----
    {
        const int rrow = threadIdx.x >> 4;
        const int sub  = threadIdx.x & 15;
        h16* pr = P + rrow * LDP + sub * 64;

        float mx = -3.0e38f;
        for (int i = 0; i < 64; ++i) mx = fmaxf(mx, (float)pr[i]);
        red[rrow][sub] = mx;
        __syncthreads();
        if (sub == 0) {
            float m2 = red[rrow][0];
            for (int i = 1; i < 16; ++i) m2 = fmaxf(m2, red[rrow][i]);
            rowstat[rrow] = m2;
        }
        __syncthreads();
        const float m2 = rowstat[rrow];

        float sum = 0.f;
        for (int i = 0; i < 64; ++i) {
            const float e = __expf((float)pr[i] - m2);
            sum += e;
            pr[i] = (h16)e;
        }
        red[rrow][sub] = sum;
        __syncthreads();
        if (sub == 0) {
            float s = 0.f;
            for (int i = 0; i < 16; ++i) s += red[rrow][i];
            rowstat[rrow] = 1.0f / s;
        }
        __syncthreads();
        const float inv = rowstat[rrow];
        for (int i = 0; i < 64; ++i) pr[i] = (h16)((float)pr[i] * inv);
    }
    __syncthreads();

    // ---- Phase 3: O^T tile = P @ Hv^T ----
    {
        v8f acc[4];
#pragma unroll
        for (int i = 0; i < 4; ++i) acc[i] = zero;
        const int c0 = wave * 64;
        const h16* Hb = Hv + (size_t)b * Cc * Tt;   // (c, s) rows
        for (int kk = 0; kk < Tt; kk += 32) {
            v16h a = load_a16x32(P + kk, LDP);      // probs from LDS
#pragma unroll
            for (int ct = 0; ct < 4; ++ct) {
                v16h bb = load_b32x16(Hb + (size_t)(c0 + ct * 16) * Tt + kk, Tt);
                acc[ct] = wmma32(a, bb, acc[ct]);
            }
        }
        const int n  = lane & 15;
        const int mb = (lane >> 4) * 8;
#pragma unroll
        for (int ct = 0; ct < 4; ++ct)
#pragma unroll
            for (int i = 0; i < 8; ++i)
                OT[((size_t)b * Tt + tq0 + mb + i) * Cc + c0 + ct * 16 + n] =
                    (h16)acc[ct][i];
    }
}

// ---------------------------------------------------------------------------
// Host launcher
// ---------------------------------------------------------------------------
extern "C" void kernel_launch(void* const* d_in, const int* in_sizes, int n_in,
                              void* d_out, int out_size, void* d_ws, size_t ws_size,
                              hipStream_t stream) {
    (void)in_sizes; (void)n_in; (void)out_size; (void)ws_size;

    const float* content = (const float*)d_in[0];
    const float* style   = (const float*)d_in[1];
    const float* Wf = (const float*)d_in[2];
    const float* bf = (const float*)d_in[3];
    const float* Wg = (const float*)d_in[4];
    const float* bg = (const float*)d_in[5];
    const float* Wh = (const float*)d_in[6];
    const float* bh = (const float*)d_in[7];
    const float* Wo = (const float*)d_in[8];
    const float* bo = (const float*)d_in[9];
    float* out = (float*)d_out;

    // Workspace layout (f16 elements)
    h16* ws = (h16*)d_ws;
    const size_t WSZ = (size_t)Cc * Cc;           // 262144
    const size_t TSZ = (size_t)Bn * Tt * Cc;      // 8388608
    h16* WfH = ws;                 h16* WgH = WfH + WSZ;
    h16* WhH = WgH + WSZ;          h16* WoH = WhH + WSZ;
    h16* XcT = WoH + WSZ;          // normalized content, (B,T,C)
    h16* XsT = XcT + TSZ;          // normalized style,   (B,T,C)
    h16* XsR = XsT + TSZ;          // raw style f16,      (B,T,C)
    h16* FqT = XsR + TSZ;          // (B,Tc,C)
    h16* GkT = FqT + TSZ;          // (B,Ts,C)
    h16* HvB = GkT + TSZ;          // (B,C,Ts)
    h16* OT  = HvB + TSZ;          // (B,Tc,C)

    // 1) weights -> f16
    cvt_weights_kernel<<<dim3((Cc * Cc) / 256, 4), 256, 0, stream>>>(
        Wf, Wg, Wh, Wo, WfH, WgH, WhH, WoH);

    // 2) MVN + cast/transpose
    mvn_cast_kernel<<<dim3(Bn * Cc, 2), 256, 0, stream>>>(
        content, style, XcT, XsT, XsR);

    // 3) conv1x1 GEMMs (WMMA), 32x64 wave tiles
    const int convBlocks = (Bn * (Cc / 32) * (Tt / 64)) / 8;   // 512
    conv1x1_gemm_kernel<1><<<convBlocks, 256, 0, stream>>>(WfH, bf, XcT, FqT, nullptr);
    conv1x1_gemm_kernel<1><<<convBlocks, 256, 0, stream>>>(WgH, bg, XsT, GkT, nullptr);
    conv1x1_gemm_kernel<0><<<convBlocks, 256, 0, stream>>>(WhH, bh, XsR, HvB, nullptr);

    // 4) fused attention (logits + softmax + PV), S never touches HBM
    attn_fused_kernel<<<Bn * (Tt / 16), 256, 0, stream>>>(FqT, GkT, HvB, OT);

    // 5) final conv1x1 -> fp32 output (B,C,Tc)
    conv1x1_gemm_kernel<2><<<convBlocks, 256, 0, stream>>>(WoH, bo, OT, nullptr, out);
}
